// Decoder_18416819765462
// MI455X (gfx1250) — matrix operand order repaired and checked by reading, not compiled, not run on hardware
//
#include <hip/hip_runtime.h>
#include <hip/hip_bf16.h>
#include <stdint.h>

typedef __attribute__((ext_vector_type(16))) __bf16 v16bf;
typedef __attribute__((ext_vector_type(8)))  __bf16 v8bf;
typedef __attribute__((ext_vector_type(8)))  float  v8f;

#define NWG 64
#define TT  300
#define BB  16
#define SS  160
#define ATT 128
#define ENC 512
#define DIM 1024
#define PRE 256
#define MEL_OUT_ELEMS (16*80*300)
#define STOP_ELEMS    (16*300)
#define ATTN_OFF      (MEL_OUT_ELEMS + STOP_ELEMS)

// ---------------- WMMA helpers (CDNA5 layouts, 16x16x32 bf16) ----------------

__device__ __forceinline__ v8f wmma_bf16(v16bf a, v16bf b, v8f c) {
  // D = A(16x32 bf16) * B(32x16 bf16) + C(16x16 f32)
  return __builtin_amdgcn_wmma_f32_16x16x32_bf16(false, a, false, b, (short)0, c, false, false);
}

// A-matrix 16x32 bf16 fragment, row-major source A[m][k], lda elements.
// lane L: M = L%16; half=L/16 holds K {k0+16h..k0+16h+15} (contiguous).
__device__ __forceinline__ v16bf load_afrag(const __bf16* A, int lda, int k0, int lane) {
  int m = lane & 15, half = lane >> 4;
  const __bf16* pa = A + (long)m * lda + k0 + half * 16;
  const __bf16* pb = A + (long)m * lda + k0 + 8 + half * 16;
  v8bf lo = *(const v8bf*)pa;
  v8bf hi = *(const v8bf*)pb;
  return __builtin_shufflevector(lo, hi, 0,1,2,3,4,5,6,7,8,9,10,11,12,13,14,15);
}

// A fragment where K range is split over two row-major buffers X (K<xk) and H.
__device__ __forceinline__ v16bf load_afrag2(const __bf16* X, int ldx, int xk,
                                             const __bf16* H, int ldh, int k0, int lane) {
  int m = lane & 15, half = lane >> 4;
  int ka = k0 + half * 16;
  int kb = k0 + 8 + half * 16;
  const __bf16* pa = (ka < xk) ? (X + (long)m * ldx + ka) : (H + (long)m * ldh + (ka - xk));
  const __bf16* pb = (kb < xk) ? (X + (long)m * ldx + kb) : (H + (long)m * ldh + (kb - xk));
  v8bf lo = *(const v8bf*)pa;
  v8bf hi = *(const v8bf*)pb;
  return __builtin_shufflevector(lo, hi, 0,1,2,3,4,5,6,7,8,9,10,11,12,13,14,15);
}

// LDS A fragment (same layout, shared-memory source)
__device__ __forceinline__ v16bf load_afrag_lds(const __bf16* A, int lda, int k0, int lane) {
  int m = lane & 15, half = lane >> 4;
  const __bf16* pa = A + m * lda + k0 + half * 16;
  const __bf16* pb = A + m * lda + k0 + 8 + half * 16;
  v8bf lo = *(const v8bf*)pa;
  v8bf hi = *(const v8bf*)pb;
  return __builtin_shufflevector(lo, hi, 0,1,2,3,4,5,6,7,8,9,10,11,12,13,14,15);
}

// B-matrix 32x16 bf16 fragment. W stored [Ncols][K] row-major (torch layout).
// lane L: N = L%16; half=L/16 holds K {k0+16h .. k0+16h+15} (32B contiguous).
__device__ __forceinline__ v16bf load_bfrag(const __bf16* W, int ldk, int nbase, int k0, int lane) {
  int n = lane & 15, half = lane >> 4;
  return *(const v16bf*)(W + (long)(nbase + n) * ldk + k0 + half * 16);
}

__device__ __forceinline__ float sigm(float x) { return 1.0f / (1.0f + __expf(-x)); }

// ---------------- device-wide split barrier (persistent kernel) ----------------

__device__ void grid_sync(unsigned* cnt, unsigned* gen) {
  __syncthreads();
  if (threadIdx.x == 0) {
    __threadfence();
    unsigned g = __hip_atomic_load(gen, __ATOMIC_RELAXED, __HIP_MEMORY_SCOPE_AGENT);
    unsigned a = __hip_atomic_fetch_add(cnt, 1u, __ATOMIC_ACQ_REL, __HIP_MEMORY_SCOPE_AGENT);
    if (a == (unsigned)(NWG - 1)) {
      __hip_atomic_store(cnt, 0u, __ATOMIC_RELAXED, __HIP_MEMORY_SCOPE_AGENT);
      __hip_atomic_store(gen, g + 1u, __ATOMIC_RELEASE, __HIP_MEMORY_SCOPE_AGENT);
    } else {
      while (__hip_atomic_load(gen, __ATOMIC_ACQUIRE, __HIP_MEMORY_SCOPE_AGENT) == g)
        __builtin_amdgcn_s_sleep(2);
    }
    __threadfence();
  }
  __syncthreads();
}

// ---------------- init: bf16 weight packing + state init ----------------

__global__ void init_kernel(const float* __restrict__ enc_out, const float* __restrict__ Wt,
                            const float* __restrict__ Wq, const float* __restrict__ conv2,
                            const float* __restrict__ Wih0, const float* __restrict__ Whh0,
                            const float* __restrict__ bih0, const float* __restrict__ bhh0,
                            const float* __restrict__ Wih1, const float* __restrict__ Whh1,
                            const float* __restrict__ bih1, const float* __restrict__ bhh1,
                            const float* __restrict__ h0, const float* __restrict__ c0,
                            const float* __restrict__ projW,
                            __bf16* W0, __bf16* W1, __bf16* WP, __bf16* Wenc,
                            __bf16* Wqb, __bf16* conv2b, __bf16* Aenc,
                            float* bias0, float* bias1,
                            __bf16* A0h, __bf16* A1h1,
                            float* c0s, float* c1s, float* cum, unsigned* syncp) {
  long tid = (long)blockIdx.x * blockDim.x + threadIdx.x;
  long stride = (long)gridDim.x * blockDim.x;
  for (long i = tid; i < 4096L * 1792; i += stride) {
    int o = (int)(i / 1792), k = (int)(i % 1792);
    float v = (k < 768) ? Wih0[(long)o * 768 + k] : Whh0[(long)o * 1024 + (k - 768)];
    W0[i] = (__bf16)v;
  }
  for (long i = tid; i < 4096L * 2048; i += stride) {
    int o = (int)(i / 2048), k = (int)(i % 2048);
    float v = (k < 1024) ? Wih1[(long)o * 1024 + k] : Whh1[(long)o * 1024 + (k - 1024)];
    W1[i] = (__bf16)v;
  }
  for (long i = tid; i < 96L * 1536; i += stride) {
    int o = (int)(i / 1536), k = (int)(i % 1536);
    WP[i] = (__bf16)((o < 81) ? projW[(long)o * 1536 + k] : 0.0f);
  }
  for (long i = tid; i < 128L * 512; i += stride) Wenc[i] = (__bf16)Wt[i];
  for (long i = tid; i < 128L * 1024; i += stride) Wqb[i] = (__bf16)Wq[i];
  for (long i = tid; i < 128L * 32; i += stride) conv2b[i] = (__bf16)conv2[i];
  for (long i = tid; i < (long)BB * SS * ENC; i += stride) Aenc[i] = (__bf16)enc_out[i];
  for (long i = tid; i < 4096; i += stride) {
    bias0[i] = bih0[i] + bhh0[i];
    bias1[i] = bih1[i] + bhh1[i];
  }
  for (long i = tid; i < (long)BB * DIM; i += stride) {
    int k = (int)(i & (DIM - 1));
    A0h[i]  = (__bf16)h0[k];          // parity 0 read at t=0
    A1h1[i] = (__bf16)h0[DIM + k];    // parity 0
    c0s[i]  = c0[k];
    c1s[i]  = c0[DIM + k];
  }
  for (long i = tid; i < BB * SS; i += stride) cum[i] = 0.0f;
  if (tid < 8) syncp[tid] = 0u;
}

// ---------------- prenet: per-frame 80->256->256 with ReLU, bf16 out ----------------

__global__ __launch_bounds__(256)
void prenet_kernel(const float* __restrict__ mels,
                   const float* __restrict__ pW1, const float* __restrict__ pb1,
                   const float* __restrict__ pW2, const float* __restrict__ pb2,
                   __bf16* melp) {
  int t = blockIdx.x >> 4, b = blockIdx.x & 15;
  __shared__ float sm[80];
  __shared__ float sp[256];
  if (threadIdx.x < 80)
    sm[threadIdx.x] = (t == 0) ? 0.0f : mels[((long)b * 80 + threadIdx.x) * TT + (t - 1)];
  __syncthreads();
  int o = threadIdx.x;
  float a = pb1[o];
  for (int c = 0; c < 80; ++c) a += pW1[o * 80 + c] * sm[c];
  sp[o] = fmaxf(a, 0.0f);
  __syncthreads();
  float a2 = pb2[o];
  for (int c = 0; c < 256; ++c) a2 += pW2[o * 256 + c] * sp[c];
  melp[((long)t * BB + b) * PRE + o] = (__bf16)fmaxf(a2, 0.0f);
}

// ---------------- l_enc = enc_out @ Wt^T via WMMA (2560x128, K=512) ----------------

__global__ __launch_bounds__(256)
void lenc_kernel(const __bf16* __restrict__ Aenc, const __bf16* __restrict__ Wenc,
                 float* __restrict__ lencp) {
  int mt = blockIdx.x;                 // 160 tiles of 16 rows
  int wave = threadIdx.x >> 5, lane = threadIdx.x & 31;
  const __bf16* Abase = Aenc + (long)mt * 16 * ENC;
  v8f acc = {};
  for (int ks = 0; ks < ENC / 32; ++ks) {
    v16bf a = load_afrag(Abase, ENC, ks * 32, lane);
    v16bf b = load_bfrag(Wenc, ENC, wave * 16, ks * 32, lane);
    acc = wmma_bf16(a, b, acc);
  }
  int n = lane & 15, half = lane >> 4;
  for (int r = 0; r < 8; ++r) {
    int row = mt * 16 + r + 8 * half;
    lencp[(long)row * ATT + wave * 16 + n] = acc[r];
  }
}

// ---------------- persistent decoder scan ----------------

__global__ __launch_bounds__(256)
void decoder_kernel(const float* __restrict__ enc_out, const int* __restrict__ text_len,
                    const float* __restrict__ w_score, const float* __restrict__ conv1,
                    const float* __restrict__ lencp, const __bf16* __restrict__ melp,
                    const __bf16* __restrict__ Wqb, const __bf16* __restrict__ conv2b,
                    const __bf16* __restrict__ W0, const __bf16* __restrict__ W1,
                    const float* __restrict__ bias0, const float* __restrict__ bias1,
                    __bf16* A0x, __bf16* A0h, __bf16* A1h0, __bf16* A1h1,
                    float* c0s, float* c1s, float* cum,
                    __bf16* cell, float* d_attn, unsigned* syncp) {
  __shared__ float s_part[8][16][16];            // per-wave WMMA partials (8KB)
  __shared__ __align__(16) __bf16 s_locb[SS * 32]; // location features, A-tile layout (10KB)
  __shared__ float s_q[ATT];
  __shared__ float s_cum[SS];
  __shared__ float s_score[SS];
  __shared__ float s_sc[256];
  __shared__ float s_tmp[256];

  unsigned* cnt = syncp;
  unsigned* gen = syncp + 1;
  int wave = threadIdx.x >> 5, lane = threadIdx.x & 31;
  int hbase = blockIdx.x * 16;          // 16 hidden units per WG, 64 WGs = 1024

  for (int t = 0; t < TT; ++t) {
    // ---------- phase 1: attention (WGs 0..15, one per batch) ----------
    if (blockIdx.x < BB) {
      int b = blockIdx.x;
      for (int s = threadIdx.x; s < SS; s += 256) {
        s_cum[s] = cum[b * SS + s];
        s_score[s] = 0.0f;
      }
      __syncthreads();
      // conv1 over cum (K=31, 'same'), write bf16 A-tile [s][l]
      for (int i = threadIdx.x; i < 32 * SS; i += 256) {
        int l = i / SS, s = i % SS;
        float acc = 0.0f;
        for (int k = 0; k < 31; ++k) {
          int sp = s + k - 15;
          if (sp >= 0 && sp < SS) acc += conv1[l * 31 + k] * s_cum[sp];
        }
        s_locb[s * 32 + l] = (__bf16)acc;
      }
      // q = h1 @ Wq^T via WMMA: A = bf16 h1 (parity buf), 16x128, K=1024
      {
        const __bf16* A1hp = A1h1 + (long)(t & 1) * BB * DIM;
        v8f acc = {};
        for (int ks = 0; ks < DIM / 32; ++ks) {
          v16bf a = load_afrag(A1hp, DIM, ks * 32, lane);
          v16bf bf = load_bfrag(Wqb, DIM, wave * 16, ks * 32, lane);
          acc = wmma_bf16(a, bf, acc);
        }
        int n = lane & 15, half = lane >> 4;
        for (int r = 0; r < 8; ++r)
          if (r + 8 * half == b) s_q[wave * 16 + n] = acc[r];   // keep only row b
      }
      __syncthreads();
      // loc2 = loc @ conv2^T via WMMA (160x128, K=32) fused with tanh-score
      {
        int n = lane & 15, half = lane >> 4;
        int a = wave * 16 + n;
        float wsa = w_score[a];
        float qa = s_q[a];
        v16bf bf = load_bfrag(conv2b, 32, wave * 16, 0, lane);
        for (int mt = 0; mt < SS / 16; ++mt) {
          v16bf af = load_afrag_lds(s_locb + mt * 16 * 32, 32, 0, lane);
          v8f c = {};
          c = wmma_bf16(af, bf, c);
          for (int r = 0; r < 8; ++r) {
            int s = mt * 16 + r + 8 * half;
            float v = wsa * tanhf(qa + lencp[((long)b * SS + s) * ATT + a] + c[r]);
            atomicAdd(&s_score[s], v);
          }
        }
      }
      __syncthreads();
      // masked softmax over S
      float sv = -3.0e38f;
      if (threadIdx.x < SS)
        sv = (threadIdx.x >= (unsigned)text_len[b]) ? -3.0e38f : s_score[threadIdx.x];
      s_tmp[threadIdx.x] = sv;
      __syncthreads();
      for (int off = 128; off > 0; off >>= 1) {
        if (threadIdx.x < off) s_tmp[threadIdx.x] = fmaxf(s_tmp[threadIdx.x], s_tmp[threadIdx.x + off]);
        __syncthreads();
      }
      float mx = s_tmp[0];
      __syncthreads();
      float ev = (threadIdx.x < SS) ? __expf(sv - mx) : 0.0f;
      s_tmp[threadIdx.x] = ev;
      __syncthreads();
      for (int off = 128; off > 0; off >>= 1) {
        if (threadIdx.x < off) s_tmp[threadIdx.x] += s_tmp[threadIdx.x + off];
        __syncthreads();
      }
      float inv = 1.0f / s_tmp[0];
      __syncthreads();
      float attn = ev * inv;
      if (threadIdx.x < SS) {
        s_sc[threadIdx.x] = attn;
        cum[b * SS + threadIdx.x] = s_cum[threadIdx.x] + attn;
        d_attn[((long)b * SS + threadIdx.x) * TT + t] = attn;
      }
      __syncthreads();
      for (int e = threadIdx.x; e < ENC; e += 256) {   // align = attn @ enc_out
        float acc = 0.0f;
        for (int s = 0; s < SS; ++s) acc += s_sc[s] * enc_out[((long)b * SS + s) * ENC + e];
        __bf16 ab = (__bf16)acc;
        A0x[(long)b * 768 + PRE + e] = ab;
        cell[((long)t * BB + b) * 1536 + DIM + e] = ab;
      }
      for (int i = threadIdx.x; i < PRE; i += 256)
        A0x[(long)b * 768 + i] = melp[((long)t * BB + b) * PRE + i];
    }
    grid_sync(cnt, gen);

    // ---------- phase 2: LSTM cell 0 gates via WMMA (all 64 WGs) ----------
    {
      int g = wave & 3, kh = wave >> 2;       // gate, K-half
      int nbase = g * DIM + hbase;
      int n = lane & 15, half = lane >> 4;
      const __bf16* wrow = W0 + (long)(nbase + n) * 1792 + kh * 896 + (half << 4);
      const __bf16* A0hp = A0h + (long)(t & 1) * BB * DIM;
      v8f acc = {};
      for (int ks = 0; ks < 28; ++ks) {
        __builtin_prefetch((const void*)(wrow + (ks + 2) * 32), 0, 1);
        v16bf bfr = *(const v16bf*)(wrow + ks * 32);
        v16bf afr = load_afrag2(A0x, 768, 768, A0hp, DIM, kh * 896 + ks * 32, lane);
        acc = wmma_bf16(afr, bfr, acc);
      }
      for (int r = 0; r < 8; ++r) s_part[wave][r + 8 * half][n] = acc[r];
    }
    __syncthreads();
    {
      int bb = threadIdx.x & 15, u = threadIdx.x >> 4;
      int hidx = hbase + u;
      float gi = s_part[0][bb][u] + s_part[4][bb][u] + bias0[hidx];
      float gf = s_part[1][bb][u] + s_part[5][bb][u] + bias0[DIM + hidx];
      float gg = s_part[2][bb][u] + s_part[6][bb][u] + bias0[2 * DIM + hidx];
      float go = s_part[3][bb][u] + s_part[7][bb][u] + bias0[3 * DIM + hidx];
      float cn = sigm(gf) * c0s[bb * DIM + hidx] + sigm(gi) * tanhf(gg);
      float hn = sigm(go) * tanhf(cn);
      c0s[bb * DIM + hidx] = cn;
      __bf16 hb = (__bf16)hn;
      A1h0[bb * DIM + hidx] = hb;
      A0h[(long)((t + 1) & 1) * BB * DIM + bb * DIM + hidx] = hb;
    }
    grid_sync(cnt, gen);

    // ---------- phase 3: LSTM cell 1 gates via WMMA ----------
    {
      int g = wave & 3, kh = wave >> 2;
      int nbase = g * DIM + hbase;
      int n = lane & 15, half = lane >> 4;
      const __bf16* wrow = W1 + (long)(nbase + n) * 2048 + kh * 1024 + (half << 4);
      const __bf16* A1hp = A1h1 + (long)(t & 1) * BB * DIM;
      v8f acc = {};
      for (int ks = 0; ks < 32; ++ks) {
        __builtin_prefetch((const void*)(wrow + (ks + 2) * 32), 0, 1);
        v16bf bfr = *(const v16bf*)(wrow + ks * 32);
        v16bf afr = load_afrag2(A1h0, DIM, DIM, A1hp, DIM, kh * 1024 + ks * 32, lane);
        acc = wmma_bf16(afr, bfr, acc);
      }
      for (int r = 0; r < 8; ++r) s_part[wave][r + 8 * half][n] = acc[r];
    }
    __syncthreads();
    {
      int bb = threadIdx.x & 15, u = threadIdx.x >> 4;
      int hidx = hbase + u;
      float gi = s_part[0][bb][u] + s_part[4][bb][u] + bias1[hidx];
      float gf = s_part[1][bb][u] + s_part[5][bb][u] + bias1[DIM + hidx];
      float gg = s_part[2][bb][u] + s_part[6][bb][u] + bias1[2 * DIM + hidx];
      float go = s_part[3][bb][u] + s_part[7][bb][u] + bias1[3 * DIM + hidx];
      float cn = sigm(gf) * c1s[bb * DIM + hidx] + sigm(gi) * tanhf(gg);
      float hn = sigm(go) * tanhf(cn);
      c1s[bb * DIM + hidx] = cn;
      __bf16 hb = (__bf16)hn;
      A1h1[(long)((t + 1) & 1) * BB * DIM + bb * DIM + hidx] = hb;
      cell[((long)t * BB + bb) * 1536 + hidx] = hb;
    }
    grid_sync(cnt, gen);
  }
}

// ---------------- projection: cell_out @ projW^T via WMMA ----------------

__global__ __launch_bounds__(256)
void proj_kernel(const __bf16* __restrict__ cell, const __bf16* __restrict__ WP,
                 const float* __restrict__ projb, float* __restrict__ out) {
  int t = blockIdx.x;                   // tile = one timestep, 16 batch rows
  int wave = threadIdx.x >> 5, lane = threadIdx.x & 31;
  if (wave < 6) {                        // 96 padded output columns
    const __bf16* Abase = cell + (long)t * BB * 1536;
    v8f acc = {};
    for (int ks = 0; ks < 48; ++ks) {
      v16bf a = load_afrag(Abase, 1536, ks * 32, lane);
      v16bf b = load_bfrag(WP, 1536, wave * 16, ks * 32, lane);
      acc = wmma_bf16(a, b, acc);
    }
    int n = lane & 15, half = lane >> 4;
    for (int r = 0; r < 8; ++r) {
      int bb = r + 8 * half;
      int col = wave * 16 + n;
      if (col < 80)       out[((long)bb * 80 + col) * TT + t] = acc[r] + projb[col];
      else if (col == 80) out[MEL_OUT_ELEMS + (long)bb * TT + t] = acc[r] + projb[80];
    }
  }
}

// ---------------- host launch ----------------

extern "C" void kernel_launch(void* const* d_in, const int* in_sizes, int n_in,
                              void* d_out, int out_size, void* d_ws, size_t ws_size,
                              hipStream_t stream) {
  const float* mels    = (const float*)d_in[0];
  const float* enc_out = (const float*)d_in[1];
  const int*   tlen    = (const int*)d_in[2];
  const float* Wt      = (const float*)d_in[3];
  const float* Wq      = (const float*)d_in[4];
  const float* w_score = (const float*)d_in[5];
  const float* conv1   = (const float*)d_in[6];
  const float* conv2   = (const float*)d_in[7];
  const float* pW1     = (const float*)d_in[8];
  const float* pb1     = (const float*)d_in[9];
  const float* pW2     = (const float*)d_in[10];
  const float* pb2     = (const float*)d_in[11];
  const float* Wih0    = (const float*)d_in[12];
  const float* Whh0    = (const float*)d_in[13];
  const float* bih0    = (const float*)d_in[14];
  const float* bhh0    = (const float*)d_in[15];
  const float* Wih1    = (const float*)d_in[16];
  const float* Whh1    = (const float*)d_in[17];
  const float* bih1    = (const float*)d_in[18];
  const float* bhh1    = (const float*)d_in[19];
  const float* h0      = (const float*)d_in[20];
  const float* c0      = (const float*)d_in[21];
  const float* projW   = (const float*)d_in[22];
  const float* projb   = (const float*)d_in[23];
  float* out = (float*)d_out;

  uint8_t* ws = (uint8_t*)d_ws;
  size_t off = 0;
  auto alloc = [&](size_t bytes) -> void* {
    void* p = ws + off;
    off = (off + bytes + 255) & ~(size_t)255;
    return p;
  };
  __bf16* W0     = (__bf16*)alloc(4096L * 1792 * 2);
  __bf16* W1     = (__bf16*)alloc(4096L * 2048 * 2);
  __bf16* WP     = (__bf16*)alloc(96L * 1536 * 2);
  __bf16* Wenc   = (__bf16*)alloc(128L * 512 * 2);
  __bf16* Wqb    = (__bf16*)alloc(128L * 1024 * 2);
  __bf16* conv2b = (__bf16*)alloc(128L * 32 * 2);
  __bf16* Aenc   = (__bf16*)alloc((long)BB * SS * ENC * 2);
  __bf16* melp   = (__bf16*)alloc((long)TT * BB * PRE * 2);
  float*  lenc   = (float*)alloc((long)BB * SS * ATT * 4);
  __bf16* A0x    = (__bf16*)alloc((long)BB * 768 * 2);
  __bf16* A0h    = (__bf16*)alloc(2L * BB * DIM * 2);
  __bf16* A1h0   = (__bf16*)alloc((long)BB * DIM * 2);
  __bf16* A1h1   = (__bf16*)alloc(2L * BB * DIM * 2);
  float*  c0s    = (float*)alloc((long)BB * DIM * 4);
  float*  c1s    = (float*)alloc((long)BB * DIM * 4);
  float*  cum    = (float*)alloc((long)BB * SS * 4);
  float*  b0     = (float*)alloc(4096L * 4);
  float*  b1     = (float*)alloc(4096L * 4);
  __bf16* cell   = (__bf16*)alloc((long)TT * BB * 1536 * 2);
  unsigned* syncp = (unsigned*)alloc(256);

  float* d_attn = out + ATTN_OFF;

  init_kernel<<<1024, 256, 0, stream>>>(enc_out, Wt, Wq, conv2, Wih0, Whh0, bih0, bhh0,
                                        Wih1, Whh1, bih1, bhh1, h0, c0, projW,
                                        W0, W1, WP, Wenc, Wqb, conv2b, Aenc, b0, b1,
                                        A0h, A1h1, c0s, c1s, cum, syncp);
  prenet_kernel<<<TT * BB, 256, 0, stream>>>(mels, pW1, pb1, pW2, pb2, melp);
  lenc_kernel<<<(BB * SS) / 16, 256, 0, stream>>>(Aenc, Wenc, lenc);
  decoder_kernel<<<NWG, 256, 0, stream>>>(enc_out, tlen, w_score, conv1,
                                          lenc, melp, Wqb, conv2b, W0, W1, b0, b1,
                                          A0x, A0h, A1h0, A1h1, c0s, c1s, cum,
                                          cell, d_attn, syncp);
  proj_kernel<<<TT, 256, 0, stream>>>(cell, WP, projb, out);
}
